// MeanSpikeClassifier_53429393162882
// MI455X (gfx1250) — matrix operand order repaired and checked by reading, not compiled, not run on hardware
//
#include <hip/hip_runtime.h>
#include <hip/hip_bf16.h>

typedef __attribute__((ext_vector_type(16))) _Float16 v16h;
typedef __attribute__((ext_vector_type(8)))  float    v8f;
typedef __attribute__((ext_vector_type(2)))  float    v2f;
typedef __attribute__((ext_vector_type(4)))  float    v4f;
typedef __attribute__((ext_vector_type(4)))  int      v4i;

#ifndef __has_builtin
#define __has_builtin(x) 0
#endif

#if __has_builtin(__builtin_amdgcn_wmma_f32_16x16x4_f32)
#define HAVE_WMMA_F32X4 1
#else
#define HAVE_WMMA_F32X4 0
#endif

#define HW_H   96
#define HW_W   96
#define NPIX   (HW_H * HW_W)   // 9216
#define NSTEPS 64
#define SIM_R  0.4f            // 1 - similarity_threshold
#define SIM_R2 0.16f           // SIM_R^2

// ---------------------------------------------------------------------------
// Kernel 1: per-pixel temporal mean via WMMA f16.
// out[n] = sum_k w[k] * S[k][n], w[k] = (k >= start) ? 1/(T-start) : 0.
// One wave computes a 16-pixel tile with two 16x16x32 WMMAs (K chunks 0,32).
// A = weight vector broadcast into all 16 rows; B = spike tile (f32 -> f16).
// ---------------------------------------------------------------------------
__global__ void mean_wmma_kernel(const float* __restrict__ spikes,
                                 const int* __restrict__ css,
                                 float* __restrict__ values,
                                 float* __restrict__ grid_out) {
  const int lane = threadIdx.x & 31;
  const int wave = (int)((blockIdx.x * blockDim.x + threadIdx.x) >> 5);
  const int n0 = wave * 16;
  if (n0 >= NPIX) return;  // exact grid: never taken, kept for safety (uniform per wave)

  int start = *css - 1;
  start = start < 0 ? 0 : (start > NSTEPS - 1 ? NSTEPS - 1 : start);
  const float wscale = 1.0f / (float)(NSTEPS - start);

  const int half = lane >> 4;       // 0: lanes 0-15, 1: lanes 16-31
  const int nl   = lane & 15;
  const int n    = n0 + nl;
  const int koff = half * 16;       // B rows K-offset for upper lane half

  v8f c = {};
  for (int kbase = 0; kbase < NSTEPS; kbase += 32) {
    v16h a, b;
    // A 16x32 f16 layout (matches B): VGPR v: K = 2v + 16*half {+0,+1}
    // B 32x16 f16, contiguous k order chosen here: VGPR v: rows K = 2v(+koff), 2v+1(+koff); N = lane&15.
#pragma unroll
    for (int v = 0; v < 8; ++v) {
      const int klo = 2 * v + 16 * half;
      const int kg0 = kbase + klo;
      const int kg1 = kg0 + 1;
      a[2 * v]     = (_Float16)((kg0 >= start) ? wscale : 0.0f);
      a[2 * v + 1] = (_Float16)((kg1 >= start) ? wscale : 0.0f);
      const int bk0 = kbase + 2 * v + koff;
      const int bk1 = bk0 + 1;
      b[2 * v]     = (_Float16)spikes[(size_t)bk0 * NPIX + n];
      b[2 * v + 1] = (_Float16)spikes[(size_t)bk1 * NPIX + n];
    }
    c = __builtin_amdgcn_wmma_f32_16x16x32_f16(false, a, false, b, (short)0, c,
                                               false, false);
  }
  // All 16 C rows are identical; row 0 lives in c[0] of lanes 0-15 (N = lane).
  if (lane < 16) {
    values[n0 + lane]   = c[0];
    grid_out[n0 + lane] = c[0];
  }
}

// ---------------------------------------------------------------------------
// Kernel 2: labels[i] = i
// ---------------------------------------------------------------------------
__global__ void init_labels_kernel(int* __restrict__ labels) {
  const int i = blockIdx.x * blockDim.x + threadIdx.x;
  if (i < NPIX) labels[i] = i;
}

// ---------------------------------------------------------------------------
// Kernel 3: one min-label propagation sweep.
// One wave owns a 16-row i-tile; loops over all 576 j-tiles. Tile distance
// (v_i - v_j)^2 = v_i^2 - 2*v_i*v_j + v_j^2 where the rank-1 outer product
// v_i (x) v_j comes from a K=4 f32 WMMA (only K=0 populated). Row minima are
// folded across lanes with wave32 shfl_xor within each 16-lane half.
// ---------------------------------------------------------------------------
__global__ void prop_kernel(const float* __restrict__ vals,
                            const int* __restrict__ lin,
                            int* __restrict__ lout) {
  const int lane = threadIdx.x & 31;
  const int wave = (int)((blockIdx.x * blockDim.x + threadIdx.x) >> 5);
  const int i0 = wave * 16;
  if (i0 >= NPIX) return;  // exact grid: never taken

  const int half = lane >> 4;
  const int nl   = lane & 15;

  // C/D layout: VGPR r holds row M = r + 8*half, column N = lane&15.
  float vim[8];
  int   cur[8];
#pragma unroll
  for (int r = 0; r < 8; ++r) {
    const int m = i0 + r + 8 * half;
    vim[r] = vals[m];
    cur[r] = lin[m];  // minimum(neigh_min, labels) -> seed with own label
  }

#if HAVE_WMMA_F32X4
  // A 16x4 f32 layout: lanes 0-15 hold K=0 (x) / K=1 (y); lanes 16-31 K=2/3.
  v2f a;
  a.x = (lane < 16) ? vals[i0 + lane] : 0.0f;
  a.y = 0.0f;
#endif

  for (int j0 = 0; j0 < NPIX; j0 += 16) {
    const float vjn = vals[j0 + nl];
    const int   lj  = lin[j0 + nl];
#if HAVE_WMMA_F32X4
    // B 4x16 f32 layout: lanes 0-15 hold rows K=0 (x) / K=1 (y).
    v2f b;
    b.x = (lane < 16) ? vjn : 0.0f;
    b.y = 0.0f;
    v8f cz = {};
    v8f p = __builtin_amdgcn_wmma_f32_16x16x4_f32(false, a, false, b, (short)0,
                                                  cz, false, false);
#pragma unroll
    for (int r = 0; r < 8; ++r) {
      const float d2 = vim[r] * vim[r] - 2.0f * p[r] + vjn * vjn;
      if (d2 <= SIM_R2 + 1e-6f) cur[r] = (lj < cur[r]) ? lj : cur[r];
    }
#else
#pragma unroll
    for (int r = 0; r < 8; ++r) {
      if (fabsf(vim[r] - vjn) <= SIM_R) cur[r] = (lj < cur[r]) ? lj : cur[r];
    }
#endif
  }

  // Min-reduce each row across its 16 N-lanes (masks 1,2,4,8 stay in-half).
#pragma unroll
  for (int r = 0; r < 8; ++r) {
#pragma unroll
    for (int off = 1; off < 16; off <<= 1) {
      const int o = __shfl_xor(cur[r], off, 32);
      cur[r] = (o < cur[r]) ? o : cur[r];
    }
  }
  if (nl == 0) {
#pragma unroll
    for (int r = 0; r < 8; ++r) lout[i0 + r + 8 * half] = cur[r];
  }
}

// ---------------------------------------------------------------------------
// Kernel 4: masks[c][p] = (labels[p] == c). 340 MB streaming pass: 128-bit
// label load (L2-hot, 36 KB working set) + 128-bit non-temporal store.
// Grid: 9 blocks of 256 threads per row c (9*256*4 = 9216 columns).
// Uses clang ext_vector types (HIP float4 is a class and is rejected by
// __builtin_nontemporal_store).
// ---------------------------------------------------------------------------
__global__ void mask_kernel(const int* __restrict__ labels,
                            float* __restrict__ out) {
  const int c  = (int)(blockIdx.x / 9);
  const int p4 = (int)((blockIdx.x % 9) * blockDim.x + threadIdx.x) * 4;
  const v4i l = *(const v4i*)(labels + p4);
  v4f o;
  o.x = (l.x == c) ? 1.0f : 0.0f;
  o.y = (l.y == c) ? 1.0f : 0.0f;
  o.z = (l.z == c) ? 1.0f : 0.0f;
  o.w = (l.w == c) ? 1.0f : 0.0f;
  __builtin_nontemporal_store(o, (v4f*)(out + (size_t)c * NPIX + p4));
}

// ---------------------------------------------------------------------------
extern "C" void kernel_launch(void* const* d_in, const int* in_sizes, int n_in,
                              void* d_out, int out_size, void* d_ws, size_t ws_size,
                              hipStream_t stream) {
  (void)in_sizes; (void)n_in; (void)out_size; (void)ws_size;

  const float* spikes = (const float*)d_in[0];
  const int*   css    = (const int*)d_in[1];   // classifier_start_step (device scalar)

  float* out      = (float*)d_out;
  float* grid_out = out + (size_t)NPIX * NPIX;  // second output: 96x96 mean grid

  // Workspace layout: values f32 [NPIX] | labels0 i32 [NPIX] | labels1 i32 [NPIX]
  float* values  = (float*)d_ws;
  int*   labels0 = (int*)((char*)d_ws + (size_t)NPIX * sizeof(float));
  int*   labels1 = labels0 + NPIX;

  // 1) temporal means (576 waves, 8 waves/block)
  mean_wmma_kernel<<<dim3(NPIX / 16 / 8), dim3(256), 0, stream>>>(spikes, css,
                                                                  values, grid_out);
  // 2) labels = iota
  init_labels_kernel<<<dim3((NPIX + 255) / 256), dim3(256), 0, stream>>>(labels0);

  // 3) fixed 6 sweeps (graph diameter <= ceil(1/0.4) = 3 since v in [0,1];
  //    6 sweeps provably reach the reference while-loop fixed point)
  int* bufs[2] = {labels0, labels1};
  for (int it = 0; it < 6; ++it) {
    prop_kernel<<<dim3(NPIX / 16 / 8), dim3(256), 0, stream>>>(values, bufs[it & 1],
                                                               bufs[(it + 1) & 1]);
  }
  // 6 sweeps (even) -> final labels back in labels0.

  // 4) stream out the 9216x9216 mask matrix (bandwidth floor ~15 us)
  mask_kernel<<<dim3(9 * NPIX), dim3(256), 0, stream>>>(bufs[0], out);
}
